// PatchEmbed_29368986370389
// MI455X (gfx1250) — matrix operand order repaired and checked by reading, not compiled, not run on hardware
//
#include <hip/hip_runtime.h>
#include <hip/hip_bf16.h>

// MI455X / gfx1250 PatchEmbed: bf16 WMMA pipeline, BN folded into weights.
// prep_kernel: fp32 -> bf16 weight conversion (+BN fold) into d_ws (~967KB).
// patch_main : 1 workgroup (16 wave32) per 4 groups (N=128 cols), all
//              activations LDS-resident, v_wmma_f32_16x16x32_bf16 everywhere.
//              Input points staged via async global->LDS; weights prefetched.

typedef __attribute__((ext_vector_type(16))) __bf16 v16bf;
typedef __attribute__((ext_vector_type(8)))  __bf16 v8bf;
typedef __attribute__((ext_vector_type(8)))  float  v8f;

#define EPSF 1e-5f

// ---- workspace layout (byte offsets) ----
#define WS_W2B 0          // bf16 256x128  ->  65536 B
#define WS_W3B 65536      // bf16 512x512  -> 524288 B (BN-folded)
#define WS_W4B 589824     // bf16 384x512  -> 393216 B
#define WS_F   983040     // f32 consts (1664 floats)
#define OFF_W1F 0         // W1f[128*3] (BN-folded)
#define OFF_B1F 384       // b1f[128]
#define OFF_B2  512       // b2 [256]
#define OFF_B3  768       // b3f[512] (BN-folded)
#define OFF_B4  1280      // b4 [384]
#define NCONST  1664

union ABu { v16bf v; v8bf h[2]; };

// A-matrix 16x32 bf16 fragment (ISA 7.12.2): lane m = lane&15, hs = lane>>4,
// halves K = [8hs..8hs+7] (v0-3) and [16+8hs..16+8hs+7] (v4-7).
__device__ __forceinline__ v16bf load_a(const __bf16* __restrict__ W, int ld,
                                        int m0, int k0, int lane) {
  const __bf16* p = W + (m0 + (lane & 15)) * ld + k0 + ((lane >> 4) << 3);
  ABu u;
  u.h[0] = *(const v8bf*)p;
  u.h[1] = *(const v8bf*)(p + 16);
  return u.v;
}

// B-matrix 32x16 bf16 fragment: lane n = lane&15, hs = lane>>4,
// halves K = [8hs..8hs+7] and [16+8hs..16+8hs+7] (two ds_load_b128).
__device__ __forceinline__ v16bf load_b(const __bf16* S, int ld,
                                        int n0, int k0, int lane) {
  const __bf16* p = S + (n0 + (lane & 15)) * ld + k0 + ((lane >> 4) << 3);
  ABu u;
  u.h[0] = *(const v8bf*)p;
  u.h[1] = *(const v8bf*)(p + 16);
  return u.v;
}

#define WMMA_BF16(acc, a, b) \
  acc = __builtin_amdgcn_wmma_f32_16x16x32_bf16(false, a, false, b, (short)0, acc, false, false)

// order-preserving f32 <-> u32 map so group-max can use ds_max_u32
__device__ __forceinline__ unsigned fmap(float x) {
  unsigned b = __float_as_uint(x);
  return (b & 0x80000000u) ? ~b : (b | 0x80000000u);
}
__device__ __forceinline__ float funmap(unsigned u) {
  unsigned b = (u & 0x80000000u) ? (u ^ 0x80000000u) : ~u;
  return __uint_as_float(b);
}

__global__ void prep_kernel(
    const float* __restrict__ W1, const float* __restrict__ b1,
    const float* __restrict__ g1, const float* __restrict__ be1,
    const float* __restrict__ m1, const float* __restrict__ v1,
    const float* __restrict__ W2, const float* __restrict__ b2,
    const float* __restrict__ W3, const float* __restrict__ b3,
    const float* __restrict__ g3, const float* __restrict__ be3,
    const float* __restrict__ m3, const float* __restrict__ v3,
    const float* __restrict__ W4, const float* __restrict__ b4,
    __bf16* __restrict__ W2b, __bf16* __restrict__ W3b,
    __bf16* __restrict__ W4b, float* __restrict__ wsF) {
  int i = blockIdx.x * blockDim.x + threadIdx.x;
  if (i < 256 * 128) W2b[i] = (__bf16)W2[i];
  if (i < 384 * 512) W4b[i] = (__bf16)W4[i];
  if (i < 512 * 512) {
    int o = i >> 9;
    float inv = g3[o] * rsqrtf(v3[o] + EPSF);
    W3b[i] = (__bf16)(W3[i] * inv);
  }
  if (i < 384) {
    int c = i / 3;
    wsF[OFF_W1F + i] = W1[i] * (g1[c] * rsqrtf(v1[c] + EPSF));
  }
  if (i < 128) {
    float inv = g1[i] * rsqrtf(v1[i] + EPSF);
    wsF[OFF_B1F + i] = b1[i] * inv + be1[i] - m1[i] * inv;
  }
  if (i < 256) wsF[OFF_B2 + i] = b2[i];
  if (i < 512) {
    float inv = g3[i] * rsqrtf(v3[i] + EPSF);
    wsF[OFF_B3 + i] = b3[i] * inv + be3[i] - m3[i] * inv;
  }
  if (i < 384) wsF[OFF_B4 + i] = b4[i];
}

__global__ void __launch_bounds__(512, 1)
patch_main(const float* __restrict__ pg, const __bf16* __restrict__ W2b,
           const __bf16* __restrict__ W3b, const __bf16* __restrict__ W4b,
           const float* __restrict__ wsF, float* __restrict__ out) {
  // activations stored [column][channel] with padded K-stride (16B aligned,
  // 4-dword bank skew) so B-fragments are two conflict-free ds_load_b128
  __shared__ __align__(16) __bf16 sH [128 * 136];  //  34.0 KB  h   (K=128)
  __shared__ __align__(16) __bf16 sPF[128 * 264];  //  66.0 KB  pf  (K=256)
  __shared__ __align__(16) __bf16 sH2[128 * 520];  // 130.0 KB  h2  (K=512)
  __shared__ __align__(16) __bf16 sGF[16 * 264];   //   8.3 KB  group-max of pf
  __shared__ float    sGC[512 * 4];                //   8.0 KB  W3[:, :256]@gf + b3f
  __shared__ unsigned sMax[4 * 384];               //   6.0 KB  mapped group max
  __shared__ float    sX[4 * 96];                  //   1.5 KB  input points
  __shared__ float    sC[NCONST];                  //   6.5 KB  folded consts

  const int tid  = threadIdx.x;
  const int wave = tid >> 5;
  const int lane = tid & 31;
  const int n15  = lane & 15;
  const int hs   = lane >> 4;

  // ---- stage 0a: async global->LDS copy of this chunk's 384 input floats ----
  if (tid < 384) {
    unsigned ldsa = (unsigned)(uintptr_t)(&sX[tid]);
    const float* gp = pg + blockIdx.x * 384 + tid;
    asm volatile("global_load_async_to_lds_b32 %0, %1, off"
                 :: "v"(ldsa), "v"(gp) : "memory");
  }

  // ---- stage 0b: warm WGP$/L2 for this wave's first weight fragments ----
  {
    int mrow = (wave >> 3) * 16 + n15;  // first A-fragment rows this wave touches
    __builtin_prefetch(&W2b[mrow * 128], 0, 0);
    __builtin_prefetch(&W3b[mrow * 512], 0, 0);
    __builtin_prefetch(&W4b[mrow * 512], 0, 0);
  }

  // ---- stage 0c: constants + max-accumulator init ----
  for (int i = tid; i < NCONST; i += 512) sC[i] = wsF[i];
  for (int i = tid; i < 4 * 384; i += 512) sMax[i] = 0u;
  asm volatile("s_wait_asynccnt 0x0" ::: "memory");
  __syncthreads();

  // ---- stage 1 (VALU, tiny): h = relu(W1f @ x + b1f), [col][c] ld=136 ----
  for (int idx = tid; idx < 128 * 128; idx += 512) {
    int col = idx & 127, c = idx >> 7;
    const float* xp = &sX[(col >> 5) * 96 + (col & 31) * 3];
    float v = sC[OFF_B1F + c] + sC[OFF_W1F + c * 3] * xp[0]
            + sC[OFF_W1F + c * 3 + 1] * xp[1]
            + sC[OFF_W1F + c * 3 + 2] * xp[2];
    sH[col * 136 + c] = (__bf16)fmaxf(v, 0.f);
  }
  __syncthreads();

  // ---- stage 2: pf = W2 @ h + b2  (M=256,K=128,N=128) ----
  for (int t = wave; t < 128; t += 16) {
    int mi = t >> 3, ni = t & 7;
    v8f acc = {};
#pragma unroll
    for (int kk = 0; kk < 128; kk += 32) {
      v16bf a = load_a(W2b, 128, mi * 16, kk, lane);
      v16bf b = load_b(sH, 136, ni * 16, kk, lane);
      WMMA_BF16(acc, a, b);
    }
    int n = ni * 16 + n15, mb = mi * 16 + hs * 8;
#pragma unroll
    for (int r = 0; r < 8; r++)
      sPF[n * 264 + mb + r] = (__bf16)(acc[r] + sC[OFF_B2 + mb + r]);
  }
  __syncthreads();

  // ---- stage 3a: gf[g][m] = max over 32 cols of pf ----
  for (int i = tid; i < 256 * 4; i += 512) {
    int m = i & 255, g = i >> 8;
    float mx = -3.0e38f;
#pragma unroll 4
    for (int c = 0; c < 32; c++)
      mx = fmaxf(mx, (float)sPF[(g * 32 + c) * 264 + m]);
    sGF[g * 264 + m] = (__bf16)mx;  // cols 4..15 unused/garbage, discarded
  }
  __syncthreads();

  // ---- stage 3b: gc = W3[:, 0:256] @ gf + b3f  (broadcast half of L3) ----
  for (int t = wave; t < 32; t += 16) {
    v8f acc = {};
#pragma unroll 4
    for (int kk = 0; kk < 256; kk += 32) {
      v16bf a = load_a(W3b, 512, t * 16, kk, lane);
      v16bf b = load_b(sGF, 264, 0, kk, lane);
      WMMA_BF16(acc, a, b);
    }
    int mb = t * 16 + hs * 8;
    if (n15 < 4) {
#pragma unroll
      for (int r = 0; r < 8; r++)
        sGC[(mb + r) * 4 + n15] = acc[r] + sC[OFF_B3 + mb + r];
    }
  }
  __syncthreads();

  // ---- stage 3c: h2 = relu(W3[:, 256:512] @ pf + gc)  (M=512,K=256,N=128) ----
  for (int t = wave; t < 256; t += 16) {
    int mi = t >> 3, ni = t & 7;
    v8f acc = {};
#pragma unroll 4
    for (int kk = 0; kk < 256; kk += 32) {
      v16bf a = load_a(W3b, 512, mi * 16, 256 + kk, lane);
      v16bf b = load_b(sPF, 264, ni * 16, kk, lane);
      WMMA_BF16(acc, a, b);
    }
    int n = ni * 16 + n15, g = n >> 5, mb = mi * 16 + hs * 8;
#pragma unroll
    for (int r = 0; r < 8; r++) {
      float v = acc[r] + sGC[(mb + r) * 4 + g];
      sH2[n * 520 + mb + r] = (__bf16)fmaxf(v, 0.f);
    }
  }
  __syncthreads();

  // ---- stage 4: o = W4 @ h2 + b4 ; per-group max via ds_max_u32 ----
  for (int t = wave; t < 192; t += 16) {
    int mi = t >> 3, ni = t & 7;
    v8f acc = {};
#pragma unroll 4
    for (int kk = 0; kk < 512; kk += 32) {
      v16bf a = load_a(W4b, 512, mi * 16, kk, lane);
      v16bf b = load_b(sH2, 520, ni * 16, kk, lane);
      WMMA_BF16(acc, a, b);
    }
    int n = ni * 16 + n15, g = n >> 5, mb = mi * 16 + hs * 8;
#pragma unroll
    for (int r = 0; r < 8; r++) {
      float v = acc[r] + sC[OFF_B4 + mb + r];
      atomicMax(&sMax[g * 384 + mb + r], fmap(v));
    }
  }
  __syncthreads();

  // ---- write (4 groups) x 384 embedding ----
  for (int i = tid; i < 4 * 384; i += 512)
    out[blockIdx.x * 1536 + i] = funmap(sMax[i]);
}

extern "C" void kernel_launch(void* const* d_in, const int* in_sizes, int n_in,
                              void* d_out, int out_size, void* d_ws, size_t ws_size,
                              hipStream_t stream) {
  const float* pg  = (const float*)d_in[0];
  const float* W1  = (const float*)d_in[1];
  const float* b1  = (const float*)d_in[2];
  const float* g1  = (const float*)d_in[3];
  const float* be1 = (const float*)d_in[4];
  const float* m1  = (const float*)d_in[5];
  const float* v1  = (const float*)d_in[6];
  const float* W2  = (const float*)d_in[7];
  const float* b2  = (const float*)d_in[8];
  const float* W3  = (const float*)d_in[9];
  const float* b3  = (const float*)d_in[10];
  const float* g3  = (const float*)d_in[11];
  const float* be3 = (const float*)d_in[12];
  const float* m3  = (const float*)d_in[13];
  const float* v3  = (const float*)d_in[14];
  const float* W4  = (const float*)d_in[15];
  const float* b4  = (const float*)d_in[16];

  char* ws = (char*)d_ws;
  __bf16* W2b = (__bf16*)(ws + WS_W2B);
  __bf16* W3b = (__bf16*)(ws + WS_W3B);
  __bf16* W4b = (__bf16*)(ws + WS_W4B);
  float*  wsF = (float*)(ws + WS_F);

  prep_kernel<<<1024, 256, 0, stream>>>(W1, b1, g1, be1, m1, v1, W2, b2,
                                        W3, b3, g3, be3, m3, v3, W4, b4,
                                        W2b, W3b, W4b, wsF);
  patch_main<<<2048, 512, 0, stream>>>(pg, W2b, W3b, W4b, wsF, (float*)d_out);
}
